// EdgeSelfAttention_46411416601352
// MI455X (gfx1250) — matrix operand order repaired and checked by reading, not compiled, not run on hardware
//
#include <hip/hip_runtime.h>
#include <hip/hip_fp16.h>
#include <math.h>

typedef __attribute__((ext_vector_type(16))) _Float16 v16h;
typedef __attribute__((ext_vector_type(8)))  float    v8f;

#define NNODES 512
#define DIM    64
#define HDIM   192           // 3*D
#define NTOT   4096          // B * N_NODES
#define EDGES  131072

// LDS layout (dynamic shared):
//   b_lds: 12 ntiles * 6 ktiles * 32 lanes * 16 halves (f16)  = 73728 B
//   a_lds: 8 waves * 6 ktiles * 32 lanes * 16 halves (f16)    = 49152 B
//   s_lds: 8 waves * 16 edges * 16 cols (f32)                 =  8192 B
#define B_LDS_BYTES 73728
#define A_LDS_BYTES 49152
#define SMEM_BYTES  (B_LDS_BYTES + A_LDS_BYTES + 8192)

__global__ void __launch_bounds__(256) zero_kernel(float* __restrict__ out,
                                                   float* __restrict__ denom) {
  int i = blockIdx.x * 256 + threadIdx.x;
  if (i < NTOT * DIM) out[i] = 0.0f;
  if (i < NTOT) denom[i] = 0.0f;
}

// One block = 256 threads = 8 waves; 256 edges per block (2 tiles of 16 per wave).
__global__ void __launch_bounds__(256) attn_score_kernel(
    const float* __restrict__ nf,    // [NTOT, 64]
    const float* __restrict__ adj,   // [8, 512, 512, 64]
    const float* __restrict__ W1,    // [192, 192]
    const float* __restrict__ W2,    // [1, 192]
    const int*   __restrict__ ei,    // [2, E]
    float* __restrict__ attn,        // [E]
    float* __restrict__ denom)       // [NTOT]
{
  extern __shared__ __align__(32) char smem[];
  _Float16* b_lds = (_Float16*)smem;
  _Float16* a_lds = (_Float16*)(smem + B_LDS_BYTES);
  float*    s_lds = (float*)(smem + B_LDS_BYTES + A_LDS_BYTES);

  const int tid  = threadIdx.x;
  const int lane = tid & 31;
  const int wave = tid >> 5;
  const int half = lane >> 4;      // which 16-lane half
  const int l16  = lane & 15;

  // ---- Stage B = W1^T as pre-swizzled 16x16x32 f16 B-fragments ----
  // b_lds fragment (nt,kt), lane L, elem i  <-  W1[nt*16 + L%16][kt*32 + 16*(L/16) + i]
  for (int u = tid; u < 12 * 6 * 32; u += 256) {
    int nt  = u / 192;
    int rem = u - nt * 192;
    int kt  = rem >> 5;
    int L   = rem & 31;
    int row = nt * 16 + (L & 15);
    int col = kt * 32 + ((L >> 4) << 4);
    const float* wrow = W1 + row * HDIM + col;
    v16h b;
#pragma unroll
    for (int i = 0; i < 16; ++i) b[i] = (_Float16)wrow[i];
    *(v16h*)(b_lds + u * 16) = b;
  }
  __syncthreads();

  // per-lane W2 column values for each ntile (column n = nt*16 + l16)
  float w2v[12];
#pragma unroll
  for (int nt = 0; nt < 12; ++nt) w2v[nt] = W2[nt * 16 + l16];

  _Float16* aw = a_lds + wave * (6 * 32 * 16);
  float*    sw = s_lds + wave * 256;

  for (int it = 0; it < 2; ++it) {
    const int ebase = blockIdx.x * 256 + ((wave << 1) | it) * 16;

    // ---- Stage A: h[m][0..191] = [nf[dst] | nf[src] | adj_edge], swizzled f16 ----
    for (int m = 0; m < 16; ++m) {
      const int e  = ebase + m;
      const int s  = ei[e];
      const int d  = ei[EDGES + e];
      const int bi = s / NNODES;
      const int ls = s - bi * NNODES;
      const int ld = d - (d / NNODES) * NNODES;
      const float* nfd = nf + (size_t)d * DIM;
      const float* nfs = nf + (size_t)s * DIM;
      const float* er  = adj + (((size_t)bi * NNODES + ls) * NNODES + ld) * DIM;
      for (int k = lane; k < HDIM; k += 32) {
        float v = (k < DIM) ? nfd[k] : (k < 2 * DIM) ? nfs[k - DIM] : er[k - 2 * DIM];
        int kt  = k / 32;
        int kk  = k & 31;
        int lhi = kk >> 4;                            // which lane-half holds this K
        int i   = kk & 15;                            // element slot in v16h
        aw[(size_t)(kt * 32 + m + 16 * lhi) * 16 + i] = (_Float16)v;
      }
    }

    // Preload the wave's 6 A fragments (A is shared across all 12 ntiles)
    v16h av[6];
#pragma unroll
    for (int kt = 0; kt < 6; ++kt)
      av[kt] = *(const v16h*)(aw + (size_t)(kt * 32 + lane) * 16);

    // ---- Z = H @ W1^T, tile by tile; GELU + W2 dot fused per ntile ----
    float partial[8];
#pragma unroll
    for (int r = 0; r < 8; ++r) partial[r] = 0.0f;

#pragma unroll
    for (int nt = 0; nt < 12; ++nt) {
      v8f c = {};
#pragma unroll
      for (int kt = 0; kt < 6; ++kt) {
        v16h b = *(const v16h*)(b_lds + (size_t)((nt * 6 + kt) * 32 + lane) * 16);
        c = __builtin_amdgcn_wmma_f32_16x16x32_f16(
                false, av[kt], false, b, (short)0, c, false, false);
      }
      const float w2 = w2v[nt];
#pragma unroll
      for (int r = 0; r < 8; ++r) {
        float z = c[r];
        float g = 0.5f * z * (1.0f + erff(z * 0.70710678118654752f)); // exact GELU
        partial[r] += g * w2;
      }
    }

    // c[r] on lane L corresponds to edge m = r + 8*half, column l16:
    // reduce the 16 column-partials per edge through LDS.
#pragma unroll
    for (int r = 0; r < 8; ++r)
      sw[(r + 8 * half) * 16 + l16] = partial[r];

    if (lane < 16) {
      float s = 0.0f;
#pragma unroll
      for (int j = 0; j < 16; ++j) s += sw[lane * 16 + j];
      const int e    = ebase + lane;
      const int sidx = ei[e];
      const int didx = ei[EDGES + e];
      int diff = sidx - didx; if (diff < 0) diff = -diff;
      float sc = (diff > 1) ? expf(s) : 0.0f;
      attn[e] = sc;
      atomicAdd(&denom[didx], sc);
    }
  }
}

// One wave per edge: out[dst] += edge_row * attn/denom
__global__ void __launch_bounds__(256) aggregate_kernel(
    const float* __restrict__ adj,
    const int*   __restrict__ ei,
    const float* __restrict__ attn,
    const float* __restrict__ denom,
    float* __restrict__ out)
{
  const int e    = blockIdx.x * 8 + (threadIdx.x >> 5);
  const int lane = threadIdx.x & 31;
  const int s = ei[e];
  const int d = ei[EDGES + e];
  float a  = attn[e];
  float dn = denom[d];
  a = a / ((dn == 0.0f) ? 1.0f : dn);
  if (a != 0.0f) {
    const int bi = s / NNODES;
    const int ls = s - bi * NNODES;
    const int ld = d - (d / NNODES) * NNODES;
    const float* er = adj + (((size_t)bi * NNODES + ls) * NNODES + ld) * DIM;
    atomicAdd(&out[(size_t)d * DIM + lane],      er[lane]      * a);
    atomicAdd(&out[(size_t)d * DIM + lane + 32], er[lane + 32] * a);
  }
}

extern "C" void kernel_launch(void* const* d_in, const int* in_sizes, int n_in,
                              void* d_out, int out_size, void* d_ws, size_t ws_size,
                              hipStream_t stream) {
  const float* nf  = (const float*)d_in[0];
  const float* adj = (const float*)d_in[1];
  const float* W1  = (const float*)d_in[2];
  const float* W2  = (const float*)d_in[3];
  const int*   ei  = (const int*)d_in[4];
  float* out = (float*)d_out;

  float* attn  = (float*)d_ws;          // E floats
  float* denom = attn + EDGES;          // NTOT floats

  zero_kernel<<<(NTOT * DIM + 255) / 256, 256, 0, stream>>>(out, denom);

  attn_score_kernel<<<EDGES / 256, 256, SMEM_BYTES, stream>>>(
      nf, adj, W1, W2, ei, attn, denom);

  aggregate_kernel<<<EDGES / 8, 256, 0, stream>>>(adj, ei, attn, denom, out);
}
